// GLU_26293789786268
// MI455X (gfx1250) — compile-verified, matrix operand order checked by reading, not run on hardware
//
#include <hip/hip_runtime.h>

typedef _Float16 h16 __attribute__((ext_vector_type(16)));
typedef _Float16 h8  __attribute__((ext_vector_type(8)));
typedef float    f8  __attribute__((ext_vector_type(8)));
typedef float    f4  __attribute__((ext_vector_type(4)));

// workspace layout (in _Float16 units)
#define WS_W1LO 0               // 4 stacks * 32 mats * 2 nh * 32 lanes * 16 = 131072
#define WS_W1HI 131072          // same size
#define WS_W2LO 262144          // 64 mats * 2 * 512 = 65536
#define WS_W2HI 327680          // 32 mats * 2 kc * 2 nh * 512 = 65536
#define WS_H    393216          // 16384 * 2048 halves

// ---- CDNA5 async LDS->global store helpers (ASYNCcnt-tracked) -------------
__device__ __forceinline__ unsigned lds_off(const void* p) {
  // generic pointers to LDS: low 32 bits are the LDS byte offset
  return (unsigned)(unsigned long long)p;
}
__device__ __forceinline__ void async_store_b128(void* gptr, unsigned lds_byte) {
  unsigned long long ga = (unsigned long long)gptr;
  asm volatile("global_store_async_from_lds_b128 %0, %1, off"
               :: "v"(ga), "v"(lds_byte) : "memory");
}
__device__ __forceinline__ void wait_ds0() {
  asm volatile("s_wait_dscnt 0x0" ::: "memory");
}
__device__ __forceinline__ void wait_async0() {
  asm volatile("s_wait_asynccnt 0x0" ::: "memory");
}

// ---------------------------------------------------------------------------
// Densify layer-1 butterfly groups into 32x32 block matrices, packed into the
// WMMA B-operand lane layout (lane<16: N=lane, K in {0..7,16..23};
// lane>=16: N=lane-16, K in {8..15,24..31}).
// grid 256 x 32: blocks 0..127 -> low group (stages 0..4, contiguous blocks),
// blocks 128..255 -> high group (stages 5..9, stride-32 classes).
// ---------------------------------------------------------------------------
__global__ __launch_bounds__(32) void k_dense1(const float* __restrict__ tw,
                                               _Float16* __restrict__ w1lo,
                                               _Float16* __restrict__ w1hi) {
  __shared__ float sm[32 * 33];
  const int lane = threadIdx.x;              // column of the block matrix
  const int bi = blockIdx.x;
  const bool hig = bi >= 128;
  const int stk = (bi & 127) >> 5;
  const int blk = bi & 31;                   // block index (low) or class c (high)

  float v[32];
#pragma unroll
  for (int i = 0; i < 32; ++i) v[i] = (i == lane) ? 1.0f : 0.0f;

#pragma unroll
  for (int k = 0; k < 5; ++k) {              // local log-stride
    const int stage = hig ? (5 + k) : k;
#pragma unroll
    for (int q = 0; q < 16; ++q) {           // pair index within 32
      const int p0 = ((q >> k) << (k + 1)) | (q & ((1 << k) - 1));
      const int p1 = p0 + (1 << k);
      const int flat = hig ? (q * 32 + blk) : (blk * 16 + q);
      const float* t = tw + (((stk * 10 + stage) * 512) + flat) * 4;
      const float x0 = v[p0], x1 = v[p1];
      v[p0] = t[0] * x0 + t[1] * x1;
      v[p1] = t[2] * x0 + t[3] * x1;
    }
  }

#pragma unroll
  for (int i = 0; i < 32; ++i) sm[i * 33 + lane] = v[i];  // sm[n][k] = M[n][k] (col=lane)
  __syncthreads();

  const int l15 = lane & 15;
  const int kb = (lane < 16) ? 0 : 8;
  _Float16* dst = hig ? w1hi : w1lo;
  const int mat = stk * 32 + blk;
#pragma unroll
  for (int nh = 0; nh < 2; ++nh) {
    h16 pk;
#pragma unroll
    for (int t = 0; t < 8; ++t) {
      pk[t]     = (_Float16)sm[(nh * 16 + l15) * 33 + (kb + t)];
      pk[8 + t] = (_Float16)sm[(nh * 16 + l15) * 33 + (kb + 16 + t)];
    }
    *(h16*)(dst + ((mat * 2 + nh) * 32 + lane) * 16) = pk;
  }
}

// Layer-2 low group: 64 contiguous 32x32 blocks (stages 0..4 of 11). grid 64 x 32.
__global__ __launch_bounds__(32) void k_dense2lo(const float* __restrict__ tw,
                                                 _Float16* __restrict__ w2lo) {
  __shared__ float sm[32 * 33];
  const int lane = threadIdx.x;
  const int blk = blockIdx.x;                // 0..63
  float v[32];
#pragma unroll
  for (int i = 0; i < 32; ++i) v[i] = (i == lane) ? 1.0f : 0.0f;
#pragma unroll
  for (int k = 0; k < 5; ++k) {
#pragma unroll
    for (int q = 0; q < 16; ++q) {
      const int p0 = ((q >> k) << (k + 1)) | (q & ((1 << k) - 1));
      const int p1 = p0 + (1 << k);
      const int flat = blk * 16 + q;
      const float* t = tw + ((k * 1024) + flat) * 4;
      const float x0 = v[p0], x1 = v[p1];
      v[p0] = t[0] * x0 + t[1] * x1;
      v[p1] = t[2] * x0 + t[3] * x1;
    }
  }
#pragma unroll
  for (int i = 0; i < 32; ++i) sm[i * 33 + lane] = v[i];
  __syncthreads();
  const int l15 = lane & 15;
  const int kb = (lane < 16) ? 0 : 8;
#pragma unroll
  for (int nh = 0; nh < 2; ++nh) {
    h16 pk;
#pragma unroll
    for (int t = 0; t < 8; ++t) {
      pk[t]     = (_Float16)sm[(nh * 16 + l15) * 33 + (kb + t)];
      pk[8 + t] = (_Float16)sm[(nh * 16 + l15) * 33 + (kb + 16 + t)];
    }
    *(h16*)(w2lo + ((blk * 2 + nh) * 32 + lane) * 16) = pk;
  }
}

// Layer-2 high group: 32 stride-32 classes of 64x64 (stages 5..10). grid 32 x 64.
__global__ __launch_bounds__(64) void k_dense2hi(const float* __restrict__ tw,
                                                 _Float16* __restrict__ w2hi) {
  __shared__ float sm[64 * 65];
  const int tid = threadIdx.x;               // column 0..63
  const int c = blockIdx.x;                  // class 0..31
  float v[64];
#pragma unroll
  for (int i = 0; i < 64; ++i) v[i] = (i == tid) ? 1.0f : 0.0f;
#pragma unroll
  for (int k = 0; k < 6; ++k) {
#pragma unroll
    for (int q = 0; q < 32; ++q) {
      const int p0 = ((q >> k) << (k + 1)) | (q & ((1 << k) - 1));
      const int p1 = p0 + (1 << k);
      const int flat = q * 32 + c;
      const float* t = tw + (((5 + k) * 1024) + flat) * 4;
      const float x0 = v[p0], x1 = v[p1];
      v[p0] = t[0] * x0 + t[1] * x1;
      v[p1] = t[2] * x0 + t[3] * x1;
    }
  }
#pragma unroll
  for (int i = 0; i < 64; ++i) sm[i * 65 + tid] = v[i];
  __syncthreads();
  const int lane = tid & 31;
  const int l15 = lane & 15;
  const int kb = (lane < 16) ? 0 : 8;
  const int nh = tid >> 5;                   // only N-halves 0,1 ever needed (out truncates)
#pragma unroll
  for (int kc = 0; kc < 2; ++kc) {
    h16 pk;
#pragma unroll
    for (int t = 0; t < 8; ++t) {
      pk[t]     = (_Float16)sm[(nh * 16 + l15) * 65 + kc * 32 + kb + t];
      pk[8 + t] = (_Float16)sm[(nh * 16 + l15) * 65 + kc * 32 + kb + 16 + t];
    }
    *(h16*)(w2hi + (((c * 2 + kc) * 2 + nh) * 32 + lane) * 16) = pk;
  }
}

// ---------------------------------------------------------------------------
// Layer 1 (butterfly x4 stacks) + bias + GLU, fused. One wave = 16 rows.
// WG = 4 waves (64 rows). LDS per wave: mid[16x1032] + gbuf[16x1032] f16.
// ---------------------------------------------------------------------------
__global__ __launch_bounds__(128) void k_layer1_glu(const float* __restrict__ x,
                                                    const float* __restrict__ b1,
                                                    const _Float16* __restrict__ w1lo,
                                                    const _Float16* __restrict__ w1hi,
                                                    _Float16* __restrict__ h) {
  __shared__ _Float16 smem[4][2][16 * 1032];
  const int lane = threadIdx.x & 31;
  const int wid = threadIdx.x >> 5;
  _Float16* mid = &smem[wid][0][0];
  _Float16* gbuf = &smem[wid][1][0];
  const int row0 = (blockIdx.x * 4 + wid) * 16;
  const int l15 = lane & 15;
  const int kb = (lane < 16) ? 0 : 8;
  const int rbase = (lane < 16) ? 0 : 8;
  const float* xrow = x + (size_t)(row0 + l15) * 1024;

  for (int sp = 0; sp < 2; ++sp) {
    for (int phase = 0; phase < 2; ++phase) {      // 0: gate stack sp+2 ; 1: "a" stack sp
      const int stk = (phase == 0) ? (sp + 2) : sp;
      wait_async0();  // mid may still be read by in-flight async stores

      // ---- low group: 32 contiguous blocks, A from global x (f32 -> f16)
      for (int b = 0; b < 32; ++b) {
        const float* p = xrow + b * 32 + kb;
        const f4 f0 = *(const f4*)(p);
        const f4 f1 = *(const f4*)(p + 4);
        const f4 f2 = *(const f4*)(p + 16);
        const f4 f3 = *(const f4*)(p + 20);
        __builtin_prefetch(w1lo + (((stk * 32 + ((b + 1) & 31)) * 2) * 32 + lane) * 16, 0, 1);
        h16 a;
#pragma unroll
        for (int t = 0; t < 4; ++t) {
          a[t] = (_Float16)f0[t];  a[4 + t] = (_Float16)f1[t];
          a[8 + t] = (_Float16)f2[t];  a[12 + t] = (_Float16)f3[t];
        }
#pragma unroll
        for (int nh = 0; nh < 2; ++nh) {
          const h16 bm = *(const h16*)(w1lo + (((stk * 32 + b) * 2 + nh) * 32 + lane) * 16);
          f8 c = {};
          c = __builtin_amdgcn_wmma_f32_16x16x32_f16(false, a, false, bm, (short)0, c, false, false);
          const int col = b * 32 + nh * 16 + l15;
#pragma unroll
          for (int t = 0; t < 8; ++t) mid[(rbase + t) * 1032 + col] = (_Float16)c[t];
        }
      }

      // ---- high group: 32 stride-32 classes, A gathered from mid
      for (int cc = 0; cc < 32; ++cc) {
        __builtin_prefetch(w1hi + (((stk * 32 + ((cc + 1) & 31)) * 2) * 32 + lane) * 16, 0, 1);
        h16 a;
#pragma unroll
        for (int t = 0; t < 8; ++t) {
          a[t]     = mid[l15 * 1032 + (kb + t) * 32 + cc];
          a[8 + t] = mid[l15 * 1032 + (kb + 16 + t) * 32 + cc];
        }
#pragma unroll
        for (int nh = 0; nh < 2; ++nh) {
          const h16 bm = *(const h16*)(w1hi + (((stk * 32 + cc) * 2 + nh) * 32 + lane) * 16);
          f8 c = {};
          c = __builtin_amdgcn_wmma_f32_16x16x32_f16(false, a, false, bm, (short)0, c, false, false);
          const int col = (nh * 16 + l15) * 32 + cc;     // 0..1023 within this stack
          const float bias = b1[stk * 1024 + col];
          if (phase == 0) {
#pragma unroll
            for (int t = 0; t < 8; ++t) {
              const float g = c[t] + bias;
              gbuf[(rbase + t) * 1032 + col] = (_Float16)(1.0f / (1.0f + __expf(-g)));
            }
          } else {
            // h = (a + bias) * sigmoid(gate); stage in-place in mid (cols == cc mod 32)
#pragma unroll
            for (int t = 0; t < 8; ++t) {
              const float av = c[t] + bias;
              const float sg = (float)gbuf[(rbase + t) * 1032 + col];
              mid[(rbase + t) * 1032 + col] = (_Float16)(av * sg);
            }
          }
        }
      }

      if (phase == 1) {  // async-stream staged h tile: LDS -> global (f16)
        wait_ds0();      // async engine must observe the DS writes
#pragma unroll
        for (int r = 0; r < 16; ++r) {
#pragma unroll
          for (int t = 0; t < 4; ++t) {
            _Float16* gp = h + (size_t)(row0 + r) * 2048 + sp * 1024 + t * 256 + lane * 8;
            async_store_b128((void*)gp, lds_off(mid + r * 1032 + t * 256 + lane * 8));
          }
        }
      }
    }
  }
  wait_async0();
}

// ---------------------------------------------------------------------------
// Layer 2: n=2048 butterfly, output truncated to 1024 cols, + b2, fp32 out.
// WG = 2 waves; per wave: mid2[16x2072] f16 + out stage [16x1036] f32.
// ---------------------------------------------------------------------------
__global__ __launch_bounds__(64) void k_layer2(const _Float16* __restrict__ h,
                                               const float* __restrict__ b2,
                                               const _Float16* __restrict__ w2lo,
                                               const _Float16* __restrict__ w2hi,
                                               float* __restrict__ out) {
  __shared__ _Float16 mid2s[2][16 * 2072];
  __shared__ float obs[2][16 * 1036];
  const int lane = threadIdx.x & 31;
  const int wid = threadIdx.x >> 5;
  _Float16* mid = &mid2s[wid][0];
  float* outb = &obs[wid][0];
  const int row0 = (blockIdx.x * 2 + wid) * 16;
  const int l15 = lane & 15;
  const int kb = (lane < 16) ? 0 : 8;
  const int rbase = (lane < 16) ? 0 : 8;
  const _Float16* hrow = h + (size_t)(row0 + l15) * 2048;

  // ---- low group: 64 contiguous 32-blocks, A straight from global h (f16)
  for (int b = 0; b < 64; ++b) {
    const h8 a0 = *(const h8*)(hrow + b * 32 + kb);
    const h8 a1 = *(const h8*)(hrow + b * 32 + kb + 16);
    __builtin_prefetch(w2lo + ((((b + 1) & 63) * 2) * 32 + lane) * 16, 0, 1);
    h16 a;
#pragma unroll
    for (int t = 0; t < 8; ++t) { a[t] = a0[t]; a[8 + t] = a1[t]; }
#pragma unroll
    for (int nh = 0; nh < 2; ++nh) {
      const h16 bm = *(const h16*)(w2lo + ((b * 2 + nh) * 32 + lane) * 16);
      f8 c = {};
      c = __builtin_amdgcn_wmma_f32_16x16x32_f16(false, a, false, bm, (short)0, c, false, false);
      const int col = b * 32 + nh * 16 + l15;
#pragma unroll
      for (int t = 0; t < 8; ++t) mid[(rbase + t) * 2072 + col] = (_Float16)c[t];
    }
  }

  // ---- high group: 32 classes of 64x64 (K=64 via 2 chunks; only out cols < 1024)
  for (int cc = 0; cc < 32; ++cc) {
    __builtin_prefetch(w2hi + (((((cc + 1) & 31) * 2) * 2) * 32 + lane) * 16, 0, 1);
    h16 aA[2];
#pragma unroll
    for (int kc = 0; kc < 2; ++kc) {
#pragma unroll
      for (int t = 0; t < 8; ++t) {
        aA[kc][t]     = mid[l15 * 2072 + kc * 1024 + (kb + t) * 32 + cc];
        aA[kc][8 + t] = mid[l15 * 2072 + kc * 1024 + (kb + 16 + t) * 32 + cc];
      }
    }
#pragma unroll
    for (int nh = 0; nh < 2; ++nh) {
      const h16 b0 = *(const h16*)(w2hi + (((cc * 2 + 0) * 2 + nh) * 32 + lane) * 16);
      const h16 b1m = *(const h16*)(w2hi + (((cc * 2 + 1) * 2 + nh) * 32 + lane) * 16);
      f8 c = {};
      c = __builtin_amdgcn_wmma_f32_16x16x32_f16(false, aA[0], false, b0, (short)0, c, false, false);
      c = __builtin_amdgcn_wmma_f32_16x16x32_f16(false, aA[1], false, b1m, (short)0, c, false, false);
      const int col = (nh * 16 + l15) * 32 + cc;   // 0..1023
      const float bias = b2[col];
#pragma unroll
      for (int t = 0; t < 8; ++t) outb[(rbase + t) * 1036 + col] = c[t] + bias;
    }
  }

  // ---- async-stream fp32 output tile: LDS -> global
  wait_ds0();
#pragma unroll
  for (int r = 0; r < 16; ++r) {
#pragma unroll
    for (int t = 0; t < 8; ++t) {
      float* gp = out + (size_t)(row0 + r) * 1024 + t * 128 + lane * 4;
      async_store_b128((void*)gp, lds_off(outb + r * 1036 + t * 128 + lane * 4));
    }
  }
  wait_async0();
}

extern "C" void kernel_launch(void* const* d_in, const int* in_sizes, int n_in,
                              void* d_out, int out_size, void* d_ws, size_t ws_size,
                              hipStream_t stream) {
  (void)in_sizes; (void)n_in; (void)out_size; (void)ws_size;
  const float* x   = (const float*)d_in[0];
  const float* tw1 = (const float*)d_in[1];
  const float* b1  = (const float*)d_in[2];
  const float* tw2 = (const float*)d_in[3];
  const float* b2  = (const float*)d_in[4];
  float* out = (float*)d_out;
  _Float16* ws = (_Float16*)d_ws;
  _Float16* w1lo = ws + WS_W1LO;
  _Float16* w1hi = ws + WS_W1HI;
  _Float16* w2lo = ws + WS_W2LO;
  _Float16* w2hi = ws + WS_W2HI;
  _Float16* hbuf = ws + WS_H;

  k_dense1<<<256, 32, 0, stream>>>(tw1, w1lo, w1hi);
  k_dense2lo<<<64, 32, 0, stream>>>(tw2, w2lo);
  k_dense2hi<<<32, 64, 0, stream>>>(tw2, w2hi);
  k_layer1_glu<<<256, 128, 0, stream>>>(x, b1, w1lo, w1hi, hbuf);
  k_layer2<<<512, 64, 0, stream>>>(hbuf, b2, w2lo, w2hi, out);
}
